// MaskedRNN_90907277787748
// MI455X (gfx1250) — compile-verified, matrix operand order checked by reading, not run on hardware
//
#include <hip/hip_runtime.h>

// ---------------------------------------------------------------------------
// Bidirectional masked LSTM for MI455X (gfx1250, wave32, WMMA bf16->f32).
//   B=64, T=256, D=H=1024.
//   * pack x / W_ih / W_hh to bf16 in WMMA fragment layout (once)
//   * one big WMMA GEMM precomputes gates_pre[dir][t][b][4H] (+ biases)
//   * 256 sequential step launches: h @ W_hh^T via v_wmma_f32_16x16x32_bf16
//     (fully unrolled K-loop: loads allocate straight into WMMA operands),
//     fused LSTM-cell epilogue, ping-pong h state across launches.
// ---------------------------------------------------------------------------

typedef __bf16 bf16_t;
typedef bf16_t bf16x16 __attribute__((ext_vector_type(16)));
typedef float  floatx8 __attribute__((ext_vector_type(8)));
typedef unsigned int u32x4 __attribute__((ext_vector_type(4)));
typedef unsigned int u32x2 __attribute__((ext_vector_type(2)));
typedef float  f32x4 __attribute__((ext_vector_type(4)));

#define B_  64
#define T_  256
#define D_  1024
#define H_  1024

// 32-byte WMMA operand fragment (16 bf16 per lane)
union FragB { bf16x16 v; u32x4 q[2]; };

__device__ __forceinline__ unsigned short f2bf(float f) {
    unsigned int u = __float_as_uint(f);
    u += 0x7FFFu + ((u >> 16) & 1u);          // round-to-nearest-even
    return (unsigned short)(u >> 16);
}

__device__ __forceinline__ float sigm(float x) {
    return 1.0f / (1.0f + __expf(-x));
}
__device__ __forceinline__ float tanh_fast(float x) {
    float ax = fabsf(x);
    float e  = __expf(2.0f * ax);             // +inf OK -> t -> 1
    float t  = 1.0f - 2.0f / (e + 1.0f);
    return copysignf(t, x);
}

// ---------------------------------------------------------------------------
// x [B,T,D] f32  ->  xbf [T,B,D] bf16 (time-major). 4 elems per thread.
__global__ void k_pack_x(const float* __restrict__ x,
                         unsigned short* __restrict__ xbf) {
    size_t tid = (size_t)blockIdx.x * blockDim.x + threadIdx.x;
    size_t o4  = tid * 4;                     // flat index in [T,B,D]
    int t   = (int)(o4 >> 16);                // B*D = 65536
    int rem = (int)(o4 & 65535);
    int b   = rem >> 10;
    int k   = rem & 1023;
    f32x4 v = *(const f32x4*)(x + ((size_t)b * T_ + t) * D_ + k);
    u32x2 p;
    p.x = (unsigned)f2bf(v.x) | ((unsigned)f2bf(v.y) << 16);
    p.y = (unsigned)f2bf(v.z) | ((unsigned)f2bf(v.w) << 16);
    *(u32x2*)(xbf + o4) = p;
}

// ---------------------------------------------------------------------------
// W [4096,1024] f32 -> fragment-packed bf16:  wp[(nt*32+kt)*32+lane][0..15]
// B-fragment mapping: n = nt*16 + lane%16 ; k = kt*32 + (lane/16)*8 + (i<8?i:i+8)
__global__ void k_pack_w(const float* __restrict__ W,
                         unsigned short* __restrict__ wp) {
    int tid  = blockIdx.x * blockDim.x + threadIdx.x;  // [0, 262144)
    int lane = tid & 31;
    int kt   = (tid >> 5) & 31;
    int nt   = tid >> 10;
    int n    = nt * 16 + (lane & 15);
    int kb   = kt * 32 + ((lane >> 4) << 3);
    const float* src = W + (size_t)n * 1024 + kb;
    unsigned short* dst = wp + (size_t)tid * 16;
#pragma unroll
    for (int i = 0; i < 8; ++i) dst[i]     = f2bf(src[i]);
#pragma unroll
    for (int i = 0; i < 8; ++i) dst[8 + i] = f2bf(src[16 + i]);
}

// ---------------------------------------------------------------------------
// h_state[buf0][dir][b][j] = c_state[dir][b][j] = h0[b][j]
__global__ void k_init_state(const float* __restrict__ h0,
                             float* __restrict__ h_st,
                             float* __restrict__ c_st) {
    int tid = blockIdx.x * blockDim.x + threadIdx.x;   // [0, 131072)
    int r   = tid & 65535;
    h_st[tid] = h0[r];
    c_st[tid] = h0[r];
}

// ---------------------------------------------------------------------------
// gates[dir][t][b][4H] = xbf[t] @ W_ih_dir^T + (b_ih + b_hh)
// grid (8, 1024, 2), block 256 (8 waves). Each block: one 16-row tile of the
// [T*B, 4096] output; A packed into LDS once; each wave does 4 N-tiles.
// K-loop fully unrolled so loads register-allocate into WMMA operands.
__global__ __launch_bounds__(256) void k_input_gemm(
        const unsigned short* __restrict__ xbf,
        const unsigned short* __restrict__ wih_pack,   // [2][4096*1024] bf16
        const float* __restrict__ bih_f, const float* __restrict__ bhh_f,
        const float* __restrict__ bih_b, const float* __restrict__ bhh_b,
        float* __restrict__ gates) {
    __shared__ unsigned short ldsA[16384];             // 16 rows x 1024 k, 32 KB

    const int dir   = blockIdx.z;
    const int mtile = blockIdx.y;                      // 0..1023
    const int wave  = threadIdx.x >> 5;
    const int lane  = threadIdx.x & 31;
    const int nt0   = (blockIdx.x * 8 + wave) * 4;     // 0..252

    const int row0 = mtile * 16;
    const int t    = row0 >> 6;                        // rows/64 -> timestep
    const int b0   = row0 & 63;

    // cooperative A-fragment pack (bf16 already, relayout only)
    for (int p = threadIdx.x; p < 1024; p += 256) {
        const int lp = p & 31;
        const int kt = p >> 5;
        const int ml = lp & 15;
        const int kb = kt * 32 + ((lp >> 4) << 3);
        const unsigned short* src = xbf + ((size_t)t * 64 + b0 + ml) * 1024 + kb;
        u32x4* d = (u32x4*)(ldsA + p * 16);
        d[0] = *(const u32x4*)(src);
        d[1] = *(const u32x4*)(src + 16);
    }
    __syncthreads();

    // stream base: consecutive nt tiles are 16384 u16 apart
    const unsigned short* wp0 = wih_pack + (size_t)dir * (4096u * 1024u)
                              + (size_t)nt0 * 16384 + (size_t)lane * 16;

    floatx8 acc[4] = {};
#pragma unroll
    for (int kt = 0; kt < 32; ++kt) {
        FragB a;
        const u32x4* ap = (const u32x4*)(ldsA + (kt * 32 + lane) * 16);
        a.q[0] = ap[0]; a.q[1] = ap[1];
#pragma unroll
        for (int qd = 0; qd < 4; ++qd) {
            FragB b;
            const u32x4* bp = (const u32x4*)(wp0 + (size_t)qd * 16384
                                                 + (size_t)kt * 1024);
            b.q[0] = bp[0]; b.q[1] = bp[1];
            acc[qd] = __builtin_amdgcn_wmma_f32_16x16x32_bf16(
                false, a.v, false, b.v, (short)0, acc[qd], false, false);
        }
    }

    const float* bih = dir ? bih_b : bih_f;
    const float* bhh = dir ? bhh_b : bhh_f;
    const int nl = lane & 15;
    const int mh = (lane >> 4) << 3;
    float* gbase = gates + ((size_t)(dir * T_ + t) * 64) * 4096;
#pragma unroll
    for (int qd = 0; qd < 4; ++qd) {
        const int j = (nt0 + qd) * 16 + nl;
        const float bias = bih[j] + bhh[j];
#pragma unroll
        for (int r = 0; r < 8; ++r) {
            const int brow = b0 + r + mh;
            // streamed once, consumed much later: keep out of near caches
            __builtin_nontemporal_store(acc[qd][r] + bias,
                                        gbase + (size_t)brow * 4096 + j);
        }
    }
}

// ---------------------------------------------------------------------------
// One recurrence step (both directions). grid (16, 4, 2), block 128 (4 waves)
// to spread the latency-critical chain over ~128 WGPs.
// Each wave: 16x16 tile of [64, H] for ALL FOUR gates -> fused cell epilogue.
// K-loop fully unrolled (straight-line: no operand shuffling, max load hoist).
__global__ __launch_bounds__(128) void k_step(
        const float* __restrict__ h_in,    // [2][64][1024] (read buffer)
        float* __restrict__ h_out,         // [2][64][1024] (write buffer)
        float* __restrict__ c_st,          // [2][64][1024]
        const float* __restrict__ gates,   // [2][256][64][4096]
        const unsigned short* __restrict__ whh_pack,
        const int* __restrict__ mask,      // [64][256]
        const float* __restrict__ h0,      // [64][1024]
        float* __restrict__ out,           // [256][64][2048]
        int s) {
    __shared__ unsigned short ldsA[16384];

    const int dir  = blockIdx.z;
    const int mt   = blockIdx.y;                   // 0..3 (16-row tile of B)
    const int wave = threadIdx.x >> 5;
    const int lane = threadIdx.x & 31;
    const int nt   = blockIdx.x * 4 + wave;        // 0..63 (16-col tile of H)

    const float* hsrc = h_in + (size_t)dir * 65536;

    // pack h rows (f32 -> bf16 A fragments) into LDS
    for (int p = threadIdx.x; p < 1024; p += 128) {
        const int lp = p & 31;
        const int kt = p >> 5;
        const int ml = lp & 15;
        const int kb = kt * 32 + ((lp >> 4) << 3);
        const float* src = hsrc + (size_t)(mt * 16 + ml) * 1024 + kb;
        unsigned short* d = ldsA + p * 16;
#pragma unroll
        for (int i = 0; i < 8; ++i) d[i]     = f2bf(src[i]);
#pragma unroll
        for (int i = 0; i < 8; ++i) d[8 + i] = f2bf(src[16 + i]);
    }
    __syncthreads();

    // gate g's B-fragment stream: tiles g*64+nt -> streams 1 Mi-u16 apart
    const unsigned short* wp0 = whh_pack + (size_t)dir * (4096u * 1024u)
                              + (size_t)nt * 16384 + (size_t)lane * 16;

    floatx8 acc[4] = {};                           // i, f, g, o gate tiles
#pragma unroll
    for (int kt = 0; kt < 32; ++kt) {
        FragB a;
        const u32x4* ap = (const u32x4*)(ldsA + (kt * 32 + lane) * 16);
        a.q[0] = ap[0]; a.q[1] = ap[1];
#pragma unroll
        for (int g = 0; g < 4; ++g) {
            FragB b;
            const u32x4* bp = (const u32x4*)(wp0 + (size_t)g * (64u * 16384u)
                                                 + (size_t)kt * 1024);
            b.q[0] = bp[0]; b.q[1] = bp[1];
            acc[g] = __builtin_amdgcn_wmma_f32_16x16x32_bf16(
                false, a.v, false, b.v, (short)0, acc[g], false, false);
        }
    }

    const int t  = dir ? (255 - s) : s;            // backward scans reversed
    const int nl = lane & 15;
    const int mh = (lane >> 4) << 3;
    const int j  = nt * 16 + nl;
    const float* gb = gates + ((size_t)(dir * T_ + t) * 64) * 4096;

#pragma unroll
    for (int r = 0; r < 8; ++r) {
        const int brow = mt * 16 + r + mh;
        const float* g = gb + (size_t)brow * 4096;
        float gi = g[j]        + acc[0][r];
        float gf = g[1024 + j] + acc[1][r];
        float gg = g[2048 + j] + acc[2][r];
        float go = g[3072 + j] + acc[3][r];
        const size_t sidx = (size_t)dir * 65536 + (size_t)brow * 1024 + j;
        float c_old = c_st[sidx];
        float cn = sigm(gf) * c_old + sigm(gi) * tanh_fast(gg);
        float hn = sigm(go) * tanh_fast(cn);
        float m   = (float)mask[brow * T_ + t];
        float h0v = h0[brow * 1024 + j];
        hn = hn * m + h0v * (1.0f - m);            // reset to h0 where mask==0
        cn = cn * m + h0v * (1.0f - m);
        h_out[sidx] = hn;
        c_st[sidx]  = cn;
        // never re-read by the GPU: bypass near caches
        __builtin_nontemporal_store(hn,
            out + ((size_t)s * 64 + brow) * 2048 + dir * 1024 + j);
    }
}

// ---------------------------------------------------------------------------
// hx / cx tails of d_out
__global__ void k_final(const float* __restrict__ h_fin,
                        const float* __restrict__ c_st,
                        float* __restrict__ out) {
    int tid = blockIdx.x * blockDim.x + threadIdx.x;  // [0, 131072)
    int dir = tid >> 16;
    int r   = tid & 65535;
    int b   = r >> 10;
    int j   = r & 1023;
    size_t base = (size_t)T_ * 64 * 2048;             // after output tensor
    out[base +          (size_t)b * 2048 + dir * 1024 + j] = h_fin[tid];
    out[base + 131072 + (size_t)b * 2048 + dir * 1024 + j] = c_st[tid];
}

// ---------------------------------------------------------------------------
extern "C" void kernel_launch(void* const* d_in, const int* in_sizes, int n_in,
                              void* d_out, int out_size, void* d_ws, size_t ws_size,
                              hipStream_t stream) {
    (void)in_sizes; (void)n_in; (void)out_size; (void)ws_size;
    const float* x     = (const float*)d_in[0];
    const int*   mask  = (const int*)  d_in[1];
    const float* h0    = (const float*)d_in[2];
    const float* Wih_f = (const float*)d_in[3];
    const float* Whh_f = (const float*)d_in[4];
    const float* bih_f = (const float*)d_in[5];
    const float* bhh_f = (const float*)d_in[6];
    const float* Wih_b = (const float*)d_in[7];
    const float* Whh_b = (const float*)d_in[8];
    const float* bih_b = (const float*)d_in[9];
    const float* bhh_b = (const float*)d_in[10];
    float* out = (float*)d_out;

    char* ws = (char*)d_ws;
    unsigned short* xbf      = (unsigned short*)(ws);                // 32 MB
    unsigned short* wih_pack = (unsigned short*)(ws + 33554432);     // 16 MB
    unsigned short* whh_pack = (unsigned short*)(ws + 50331648);     // 16 MB
    float*          gates    = (float*)(ws + 67108864);              // 512 MB
    float*          h_st     = (float*)(ws + 603979776);             // 1 MB (2 bufs)
    float*          c_st     = (float*)(ws + 605028352);             // 0.5 MB

    k_pack_x<<<16384, 256, 0, stream>>>(x, xbf);
    k_pack_w<<<1024, 256, 0, stream>>>(Wih_f, wih_pack);
    k_pack_w<<<1024, 256, 0, stream>>>(Wih_b, wih_pack + 4194304);
    k_pack_w<<<1024, 256, 0, stream>>>(Whh_f, whh_pack);
    k_pack_w<<<1024, 256, 0, stream>>>(Whh_b, whh_pack + 4194304);
    k_init_state<<<512, 256, 0, stream>>>(h0, h_st, c_st);

    k_input_gemm<<<dim3(8, 1024, 2), 256, 0, stream>>>(
        xbf, wih_pack, bih_f, bhh_f, bih_b, bhh_b, gates);

    for (int s = 0; s < T_; ++s) {
        const float* h_in  = h_st + (size_t)(s & 1) * 131072;
        float*       h_out = h_st + (size_t)((s + 1) & 1) * 131072;
        k_step<<<dim3(16, 4, 2), 128, 0, stream>>>(
            h_in, h_out, c_st, gates, whh_pack, mask, h0, out, s);
    }
    // after s=255 the final h lives in buffer 0
    k_final<<<512, 256, 0, stream>>>(h_st, c_st, out);
}
